// MultiheadAttention_29600914604069
// MI455X (gfx1250) — matrix operand order repaired and checked by reading, not compiled, not run on hardware
//
#include <hip/hip_runtime.h>

#define TSEQ   4096
#define DMODEL 1024
#define NHEAD  16
#define HDIM   64

typedef __attribute__((ext_vector_type(16))) __bf16 v16bf;
typedef __attribute__((ext_vector_type(8)))  __bf16 v8bf;
typedef __attribute__((ext_vector_type(4)))  __bf16 v4bf;
typedef __attribute__((ext_vector_type(8)))  float  v8f;
typedef __attribute__((ext_vector_type(4)))  float  v4f;
typedef __attribute__((ext_vector_type(4)))  unsigned int u32x4;
typedef __attribute__((ext_vector_type(8)))  int          i32x8;
typedef __attribute__((ext_vector_type(4)))  int          i32x4;

union AF { v16bf v; v8bf h[2]; };

__device__ __forceinline__ v8f wmma_bf16(const AF& a, const AF& b, v8f c) {
  return __builtin_amdgcn_wmma_f32_16x16x32_bf16(false, a.v, false, b.v,
                                                 (short)0, c, false, false);
}

__device__ __forceinline__ float half_max16(float v) {
#pragma unroll
  for (int m = 1; m < 16; m <<= 1) v = fmaxf(v, __shfl_xor(v, m, 32));
  return v;
}
__device__ __forceinline__ float half_sum16(float v) {
#pragma unroll
  for (int m = 1; m < 16; m <<= 1) v += __shfl_xor(v, m, 32);
  return v;
}

// LDS byte offset of a generic pointer into LDS: aperture keeps offset in
// addr[31:0] (ISA 10.2).
__device__ __forceinline__ unsigned lds_offset(const void* p) {
  return (unsigned)(unsigned long long)p;
}

// ---------------------------------------------------------------------------
// Tensor Data Mover: 2D tile load Global -> LDS (descriptor per ISA ch.8).
// data_size = 2B (bf16). Pads LDS dest: every 32 DWORDs (128B row) add
// 8 DWORDs (32B) -> LDS row stride 160B, matching the fragment readers.
// ---------------------------------------------------------------------------
__device__ __forceinline__ void tdm_load_2d(unsigned ldsOff, const void* gptr,
                                            unsigned tileD0, unsigned tileD1,
                                            unsigned tenD0,  unsigned tenD1,
                                            unsigned long long strideD0) {
  unsigned long long ga = (unsigned long long)gptr;
  u32x4 g0;
  g0[0] = 1u;                                        // count=1, user descriptor
  g0[1] = ldsOff;                                    // lds_addr [63:32]
  g0[2] = (unsigned)(ga & 0xffffffffu);              // global_addr lo
  g0[3] = (unsigned)((ga >> 32) & 0x01ffffffu)       // global_addr hi (57b)
        | (2u << 30);                                // type=2 ("image")
  i32x8 g1;
  g1[0] = (int)((1u << 16)                           // data_size = 2 bytes
        |       (1u << 20)                           // pad_enable
        |       (4u << 22)                           // pad_interval: 32 DWORDs
        |       (7u << 25));                         // pad_amount: 8 DWORDs
  g1[1] = (int)(tenD0 << 16);                        // tensor_dim0[15:0]
  g1[2] = (int)((tenD0 >> 16) | (tenD1 << 16));      // dim0 hi | dim1 lo
  g1[3] = (int)((tenD1 >> 16) | (tileD0 << 16));     // dim1 hi | tile_dim0
  g1[4] = (int)(tileD1 & 0xffffu);                   // tile_dim1 (tile_dim2=0)
  g1[5] = (int)(strideD0 & 0xffffffffu);             // tensor_dim0_stride lo
  g1[6] = (int)((strideD0 >> 32) & 0xffffu);         // dim0_stride hi
  g1[7] = 0;                                         // tensor_dim1_stride = 0
  i32x4 z4 = {0, 0, 0, 0};
  i32x8 z8 = {0, 0, 0, 0, 0, 0, 0, 0};
  __builtin_amdgcn_tensor_load_to_lds(g0, g1, z4, z4, z8, 0);
}

// ---------------------------------------------------------------------------
// C = X @ W^T + bias  (M=TSEQ, N=DMODEL, K=DMODEL), BF16 WMMA, F32 accum.
// MODE 0: out -> Qh  [H][T][64]  bf16
// MODE 1: out -> Kh  [H][T][64]  bf16
// MODE 2: out -> Vt  [H][64][T]  bf16 (transposed for B-fragment loads)
// MODE 3: X is bf16 attn-out; out -> f32 [T][D] (final Wo projection)
// ---------------------------------------------------------------------------
template <int MODE>
__global__ __launch_bounds__(256) void proj_gemm(const void* __restrict__ Xv,
                                                 const float* __restrict__ W,
                                                 const float* __restrict__ bias,
                                                 void* __restrict__ outB,
                                                 float* __restrict__ outF) {
  constexpr int BM = 128, BN = 64, BK = 32, LDA = 40, LDB = 40;
  __shared__ __bf16 As[BM * LDA];
  __shared__ __bf16 Bs[BN * LDB];

  const int tid  = threadIdx.x;
  const int wave = tid >> 5, lane = tid & 31;
  const int mBase = blockIdx.x * BM, nBase = blockIdx.y * BN;
  const int nlane = lane & 15;
  const int mHalf = (lane >> 4) * 8;

  const float*  Xf = (const float*)Xv;
  const __bf16* Xb = (const __bf16*)Xv;

  v8f acc[4] = {};

  for (int k0 = 0; k0 < DMODEL; k0 += BK) {
    __syncthreads();
    // ---- stage A tile (128 x 32) into LDS, converting f32 -> bf16 ----
    if (MODE == 3) {
#pragma unroll
      for (int i = 0; i < 2; ++i) {          // 512 chunks of 8 bf16
        int c = tid + i * 256;
        int row = c >> 2, cc = c & 3;
        *(v8bf*)(As + row * LDA + cc * 8) =
            *(const v8bf*)(Xb + (size_t)(mBase + row) * DMODEL + k0 + cc * 8);
      }
    } else {
#pragma unroll
      for (int i = 0; i < 4; ++i) {          // 1024 float4
        int f = tid + i * 256;
        int row = f >> 3, c4 = f & 7;
        v4f x = *(const v4f*)(Xf + (size_t)(mBase + row) * DMODEL + k0 + c4 * 4);
        v4bf y;
        y[0] = (__bf16)x[0]; y[1] = (__bf16)x[1];
        y[2] = (__bf16)x[2]; y[3] = (__bf16)x[3];
        *(v4bf*)(As + row * LDA + c4 * 4) = y;
      }
    }
    // ---- stage B tile (W rows nBase..+63, 64 x 32) ----
#pragma unroll
    for (int i = 0; i < 2; ++i) {            // 512 float4
      int f = tid + i * 256;
      int row = f >> 3, c4 = f & 7;
      v4f x = *(const v4f*)(W + (size_t)(nBase + row) * DMODEL + k0 + c4 * 4);
      v4bf y;
      y[0] = (__bf16)x[0]; y[1] = (__bf16)x[1];
      y[2] = (__bf16)x[2]; y[3] = (__bf16)x[3];
      *(v4bf*)(Bs + row * LDB + c4 * 4) = y;
    }
    __syncthreads();

    // ---- A fragment: wave's 16 rows, this 32-wide K slice ----
    const int mrow = wave * 16 + nlane;
    const int aoff = (lane < 16) ? 0 : 8;
    AF a;
    a.h[0] = *(const v8bf*)(As + mrow * LDA + aoff);
    a.h[1] = *(const v8bf*)(As + mrow * LDA + aoff + 16);

#pragma unroll
    for (int j = 0; j < 4; ++j) {
      const int brow = j * 16 + nlane;
      const int boff = (lane < 16) ? 0 : 8;
      AF b;
      b.h[0] = *(const v8bf*)(Bs + brow * LDB + boff);
      b.h[1] = *(const v8bf*)(Bs + brow * LDB + boff + 16);
      acc[j] = wmma_bf16(a, b, acc[j]);
    }
  }

  // ---- epilogue ----
  __bf16* ob = (__bf16*)outB;
#pragma unroll
  for (int j = 0; j < 4; ++j) {
    const int n  = nBase + j * 16 + nlane;
    const float bv = bias[n];
#pragma unroll
    for (int r = 0; r < 8; ++r) {
      const int m = mBase + wave * 16 + mHalf + r;
      const float cv = acc[j][r] + bv;
      if (MODE == 3) {
        outF[(size_t)m * DMODEL + n] = cv;
      } else if (MODE == 2) {
        const int hh = n >> 6, dd = n & 63;
        ob[((size_t)hh * HDIM + dd) * TSEQ + m] = (__bf16)cv;   // Vt[h][d][t]
      } else {
        const int hh = n >> 6, dd = n & 63;
        ob[((size_t)hh * TSEQ + m) * HDIM + dd] = (__bf16)cv;   // Qh/Kh[h][t][d]
      }
    }
  }
}

// ---------------------------------------------------------------------------
// Causal flash attention: block = 128 query rows of one head, 8 waves,
// each wave owns 16 rows. K/V tiles staged by the Tensor Data Mover,
// online softmax, BF16 WMMA, F32 accum.
// ---------------------------------------------------------------------------
__global__ __launch_bounds__(256) void attn_fwd(const void* __restrict__ Qv,
                                                const void* __restrict__ Kv,
                                                const void* __restrict__ Vv,
                                                void* __restrict__ Ov) {
  constexpr int BQ = 128, BKV = 64, LDS_S = 80, LDS_P = 80;
  __shared__ __bf16 Kt[BKV * LDS_S];        // [kv][d], 160B row stride
  __shared__ __bf16 Vl[HDIM * LDS_S];       // [d][kv], 160B row stride
  __shared__ __bf16 Pl[8 * 16 * LDS_P];     // per-wave P tile 16x64

  const __bf16* Qh = (const __bf16*)Qv;
  const __bf16* Kh = (const __bf16*)Kv;
  const __bf16* Vt = (const __bf16*)Vv;
  __bf16*       Oa = (__bf16*)Ov;

  const int lane = threadIdx.x & 31, wave = threadIdx.x >> 5;
  const int h = blockIdx.y;
  const int qBase = blockIdx.x * BQ;
  const int nlane = lane & 15;
  const int mHalf = (lane >> 4) * 8;
  const int aoff  = (lane < 16) ? 0 : 8;
  const int boff  = (lane < 16) ? 0 : 8;

  const unsigned ldsKt = lds_offset(Kt);
  const unsigned ldsVl = lds_offset(Vl);

  // Q fragments for this wave's 16 rows (kept in registers)
  const int qm = qBase + wave * 16 + nlane;
  const __bf16* qrow = Qh + ((size_t)h * TSEQ + qm) * HDIM;
  AF qa[2];
#pragma unroll
  for (int kd = 0; kd < 2; ++kd) {
    qa[kd].h[0] = *(const v8bf*)(qrow + kd * 32 + aoff);
    qa[kd].h[1] = *(const v8bf*)(qrow + kd * 32 + aoff + 16);
  }

  float mrow[8], lrow[8];
  v8f o[4] = {};
#pragma unroll
  for (int r = 0; r < 8; ++r) { mrow[r] = -__builtin_inff(); lrow[r] = 0.f; }

  __bf16* pw = Pl + wave * 16 * LDS_P;
  const int mrowBase = qBase + wave * 16 + mHalf;
  const int kvEnd = qBase + BQ;             // causal: keys < qBase+128

  for (int kb = 0; kb < kvEnd; kb += BKV) {
    __syncthreads();                        // prior reads done before overwrite
    // ---- TDM staging: K tile [64][64] and V tile [64][64], wave 0 only ----
    if (threadIdx.x < 32) {
      tdm_load_2d(ldsKt, Kh + ((size_t)h * TSEQ + kb) * HDIM,
                  /*tile*/ 64, 64, /*tensor*/ 64, TSEQ, /*stride0*/ HDIM);
      tdm_load_2d(ldsVl, Vt + (size_t)h * HDIM * TSEQ + kb,
                  /*tile*/ 64, 64, /*tensor*/ 64, HDIM, /*stride0*/ TSEQ);
      __builtin_amdgcn_s_wait_tensorcnt(0);
    }
    if (kb + BKV < kvEnd)                   // hint next K tile toward L2/WGP$
      __builtin_prefetch(Kh + ((size_t)h * TSEQ + kb + BKV + (threadIdx.x & 63)) * HDIM, 0, 0);
    __syncthreads();                        // tiles visible to all waves

    // ---- S = Q K^T (16 x 64) ----
    v8f s[4] = {};
#pragma unroll
    for (int j = 0; j < 4; ++j) {
      const int krow = j * 16 + nlane;
#pragma unroll
      for (int kd = 0; kd < 2; ++kd) {
        AF b;
        b.h[0] = *(const v8bf*)(Kt + krow * LDS_S + kd * 32 + boff);
        b.h[1] = *(const v8bf*)(Kt + krow * LDS_S + kd * 32 + boff + 16);
        s[j] = wmma_bf16(qa[kd], b, s[j]);
      }
    }

    // ---- scale + causal mask ----
#pragma unroll
    for (int j = 0; j < 4; ++j) {
      const int n = kb + j * 16 + nlane;
#pragma unroll
      for (int r = 0; r < 8; ++r) {
        float v = s[j][r] * 0.125f;         // 1/sqrt(64)
        s[j][r] = (n <= mrowBase + r) ? v : -__builtin_inff();
      }
    }

    // ---- online softmax update ----
    float mnew[8], alpha[8], psum[8];
#pragma unroll
    for (int r = 0; r < 8; ++r) {
      float mx = fmaxf(fmaxf(s[0][r], s[1][r]), fmaxf(s[2][r], s[3][r]));
      mx = half_max16(mx);
      mnew[r]  = fmaxf(mrow[r], mx);
      alpha[r] = __expf(mrow[r] - mnew[r]);
      psum[r]  = 0.f;
    }
#pragma unroll
    for (int j = 0; j < 4; ++j)
#pragma unroll
      for (int r = 0; r < 8; ++r) {
        float p = __expf(s[j][r] - mnew[r]);
        s[j][r] = p;
        psum[r] += p;
      }
#pragma unroll
    for (int r = 0; r < 8; ++r) {
      lrow[r] = lrow[r] * alpha[r] + half_sum16(psum[r]);
      mrow[r] = mnew[r];
    }
#pragma unroll
    for (int j = 0; j < 4; ++j)
#pragma unroll
      for (int r = 0; r < 8; ++r) o[j][r] *= alpha[r];

    // ---- P -> per-wave LDS (C-layout -> A-layout relayout) ----
#pragma unroll
    for (int j = 0; j < 4; ++j)
#pragma unroll
      for (int r = 0; r < 8; ++r)
        pw[(mHalf + r) * LDS_P + j * 16 + nlane] = (__bf16)s[j][r];
    asm volatile("s_wait_dscnt 0x0" ::: "memory");  // order LDS RAW within wave

    // ---- O += P V ----
#pragma unroll
    for (int ks = 0; ks < 2; ++ks) {
      AF pa;
      pa.h[0] = *(const v8bf*)(pw + nlane * LDS_P + ks * 32 + aoff);
      pa.h[1] = *(const v8bf*)(pw + nlane * LDS_P + ks * 32 + aoff + 16);
#pragma unroll
      for (int jd = 0; jd < 4; ++jd) {
        const int dcol = jd * 16 + nlane;
        AF vb;
        vb.h[0] = *(const v8bf*)(Vl + dcol * LDS_S + ks * 32 + boff);
        vb.h[1] = *(const v8bf*)(Vl + dcol * LDS_S + ks * 32 + boff + 16);
        o[jd] = wmma_bf16(pa, vb, o[jd]);
      }
    }
  }

  // ---- normalize + store attn output [T][DMODEL] bf16 ----
#pragma unroll
  for (int r = 0; r < 8; ++r) {
    const float inv = 1.f / lrow[r];
    const int m = qBase + wave * 16 + mHalf + r;
#pragma unroll
    for (int jd = 0; jd < 4; ++jd) {
      const int dd = jd * 16 + nlane;
      Oa[(size_t)m * DMODEL + h * HDIM + dd] = (__bf16)(o[jd][r] * inv);
    }
  }
}

// ---------------------------------------------------------------------------
extern "C" void kernel_launch(void* const* d_in, const int* in_sizes, int n_in,
                              void* d_out, int out_size, void* d_ws, size_t ws_size,
                              hipStream_t stream) {
  const float* querys = (const float*)d_in[0];
  const float* keys   = (const float*)d_in[1];
  const float* values = (const float*)d_in[2];
  const float* Wq = (const float*)d_in[3];
  const float* bq = (const float*)d_in[4];
  const float* Wk = (const float*)d_in[5];
  const float* bk = (const float*)d_in[6];
  const float* Wv = (const float*)d_in[7];
  const float* bv = (const float*)d_in[8];
  const float* Wo = (const float*)d_in[9];
  const float* bo = (const float*)d_in[10];
  float* out = (float*)d_out;

  char* ws = (char*)d_ws;
  void* Qh = (void*)(ws + (size_t)0);               // 8 MiB  [H][T][64] bf16
  void* Kh = (void*)(ws + ((size_t)8  << 20));      // 8 MiB  [H][T][64] bf16
  void* Vt = (void*)(ws + ((size_t)16 << 20));      // 8 MiB  [H][64][T] bf16
  void* Ao = (void*)(ws + ((size_t)24 << 20));      // 8 MiB  [T][D]     bf16

  dim3 gG(TSEQ / 128, DMODEL / 64);                 // 32 x 16 blocks
  dim3 gA(TSEQ / 128, NHEAD);                       // 32 x 16 blocks

  proj_gemm<0><<<gG, 256, 0, stream>>>((const void*)querys, Wq, bq, Qh, out);
  proj_gemm<1><<<gG, 256, 0, stream>>>((const void*)keys,   Wk, bk, Kh, out);
  proj_gemm<2><<<gG, 256, 0, stream>>>((const void*)values, Wv, bv, Vt, out);
  attn_fwd<<<gA, 256, 0, stream>>>(Qh, Kh, Vt, Ao);
  proj_gemm<3><<<gG, 256, 0, stream>>>((const void*)Ao, Wo, bo, Ao, out);
}
